// MultiHeadSelfAttention_17102559772933
// MI455X (gfx1250) — matrix operand order repaired and checked by reading, not compiled, not run on hardware
//
#include <hip/hip_runtime.h>
#include <hip/hip_bf16.h>

// ---------------------------------------------------------------------------
// MI455X (gfx1250) multi-head self-attention, bf16 WMMA pipeline.
// - All GEMMs: v_wmma_f32_16x16x32_bf16 (wave32, f32 accumulate)
// - Producer staging: global_load_async_to_lds_b128 (ASYNCcnt), double-buffered
//   so the DMA of tile i+1 overlaps the WMMAs of tile i (async loads complete
//   in order per wave, so s_wait_asynccnt 1 waits only for the older copy).
// ---------------------------------------------------------------------------

typedef __bf16 bf16;
typedef __attribute__((ext_vector_type(16))) __bf16 v16bf;
typedef __attribute__((ext_vector_type(8)))  __bf16 v8bf;
typedef __attribute__((ext_vector_type(4)))  __bf16 v4bf;
typedef __attribute__((ext_vector_type(8)))  float  v8f;

#define BATCH 4
#define SEQ   2048
#define EMB   1024
#define HEADS 16
#define HDIM  64
#define MROWS (BATCH * SEQ)   // 8192

static __device__ __forceinline__ v8f wmma_bf16(v16bf a, v16bf b, v8f c) {
    // D = A(16x32 bf16) * B(32x16 bf16) + C(16x16 f32)
    return __builtin_amdgcn_wmma_f32_16x16x32_bf16(
        /*neg_a=*/false, a, /*neg_b=*/false, b,
        /*c_mod=*/(short)0, c, /*reuse_a=*/false, /*reuse_b=*/false);
}

// CDNA5 async memory->LDS copy (GVS mode: SGPR base + 32-bit VGPR offset).
// Tracked by ASYNCcnt; LDS dest address is per-lane VGPR (low 32 bits of the
// generic shared pointer = LDS byte offset, per ISA 10.2 aperture layout).
static __device__ __forceinline__ void async_load_b128(unsigned lds_off,
                                                       unsigned voff,
                                                       const void* sbase) {
    asm volatile("global_load_async_to_lds_b128 %0, %1, %2"
                 :: "v"(lds_off), "v"(voff), "s"(sbase)
                 : "memory");
}
static __device__ __forceinline__ void wait_async0() {   // all copies done
    asm volatile("s_wait_asynccnt 0x0" ::: "memory");
}
static __device__ __forceinline__ void wait_async1() {   // allow 1 in flight
    asm volatile("s_wait_asynccnt 0x1" ::: "memory");
}

// A fragment: 16x32 bf16, row-major source with leading dim ld.
// lane holds row (lane&15); elem j=0..15 -> K=kA0+j (contiguous),
// kA0 = kbase + 16*(lane>=16)  (matches B fragment k order).
static __device__ __forceinline__ v16bf load_frag_a(const bf16* base, int ld,
                                                    int rowbase, int kbase, int lane) {
    int r   = rowbase + (lane & 15);
    int kA0 = kbase + ((lane >> 4) << 4);
    const bf16* p = base + (size_t)r * ld + kA0;
    v8bf lo = *(const v8bf*)(p);        // 16B load
    v8bf hi = *(const v8bf*)(p + 8);    // 16B load
    v16bf a;
#pragma unroll
    for (int i = 0; i < 8; ++i) { a[i] = lo[i]; a[i + 8] = hi[i]; }
    return a;
}

// B fragment: 32x16 bf16 from K-contiguous storage BT[col][k], leading dim ld.
// lane holds col (lane&15); elem j -> K = kbase + 16*(lane>=16) + j (contiguous).
static __device__ __forceinline__ v16bf load_frag_b(const bf16* base, int ld,
                                                    int colbase, int kbase, int lane) {
    int c  = colbase + (lane & 15);
    int kB = kbase + ((lane >> 4) << 4);
    return *(const v16bf*)(base + (size_t)c * ld + kB);   // 32B contiguous
}

// ---------------------------------------------------------------------------
// Precision conversion kernels
// ---------------------------------------------------------------------------
__global__ void cast_bf16_kernel(const float* __restrict__ x, bf16* __restrict__ y, int n4) {
    int i = blockIdx.x * blockDim.x + threadIdx.x;   // one float4 per thread
    if (i < n4) {
        float4 f = ((const float4*)x)[i];
        v4bf o; o[0] = (bf16)f.x; o[1] = (bf16)f.y; o[2] = (bf16)f.z; o[3] = (bf16)f.w;
        ((v4bf*)y)[i] = o;
    }
}

// w: [K][Nout] fp32  ->  wT: [Nout][K] bf16  (K contiguous for B fragments)
__global__ void transpose_cast_kernel(const float* __restrict__ w, bf16* __restrict__ wT,
                                      int K, int Nout) {
    int k = blockIdx.x * blockDim.x + threadIdx.x;
    int n = blockIdx.y;
    if (k < K) wT[(size_t)n * K + k] = (bf16)w[(size_t)k * Nout + n];
}

// ---------------------------------------------------------------------------
// bf16 GEMM: C[M][Nout] = A[M][K] * BT[Nout][K]^T + bias, WMMA 16x16x32.
// Block = 256 threads = 8 waves; block tile 256x64; wave tile 32x64
// (8 WMMAs per K-step per wave). B tiles async-DMA'd into double-buffered LDS.
// ---------------------------------------------------------------------------
template <bool OUT_BF16>
__global__ void __launch_bounds__(256)
gemm_bf16_kernel(const bf16* __restrict__ A, const bf16* __restrict__ BT,
                 const float* __restrict__ bias, void* __restrict__ Cout,
                 int M, int Nout, int K) {
    __shared__ __align__(16) bf16 Bs[2][64][32];   // [buf][col][k]
    const int tid  = threadIdx.x;
    const int lane = tid & 31;
    const int wave = tid >> 5;
    const int colbase = blockIdx.x * 64;
    const int rowbase = blockIdx.y * 256 + wave * 32;

    // async staging assignment: one b128 per thread per tile (64x32 = 4KB)
    const int sc  = tid >> 2;             // col 0..63
    const int skk = (tid & 3) * 8;        // k 0,8,16,24
    const unsigned lds_elem = (unsigned)(sc * 64 + skk * 2);
    const unsigned lds0 = (unsigned)(size_t)(&Bs[0][0][0]) + lds_elem;
    const unsigned lds1 = (unsigned)(size_t)(&Bs[1][0][0]) + lds_elem;
    unsigned voff = (unsigned)(((colbase + sc) * K + skk) * 2);

    // prologue: DMA tile 0 into buffer 0
    async_load_b128(lds0, voff, BT);
    voff += 64;

    v8f acc[2][4] = {};
    int p = 0;
    for (int kb = 0; kb < K; kb += 32) {
        // prefetch tile kb+32 into the alternate buffer, keep it in flight
        if (kb + 32 < K) {
            async_load_b128(p ? lds0 : lds1, voff, BT);
            voff += 64;
            wait_async1();          // wait only for tile kb (older, in-order)
        } else {
            wait_async0();
        }

        if (kb + 32 < K)   // hint next A strip into cache (global_prefetch_b8)
            __builtin_prefetch(A + (size_t)(rowbase + (lane & 15)) * K + kb + 32, 0, 1);

        v16bf a0 = load_frag_a(A, K, rowbase,      kb, lane);
        v16bf a1 = load_frag_a(A, K, rowbase + 16, kb, lane);

        __syncthreads();            // tile kb visible from all waves
#pragma unroll
        for (int t = 0; t < 4; ++t) {
            v16bf b = load_frag_b(&Bs[p][0][0], 32, t * 16, 0, lane);
            acc[0][t] = wmma_bf16(a0, b, acc[0][t]);
            acc[1][t] = wmma_bf16(a1, b, acc[1][t]);
        }
        __syncthreads();            // reads of Bs[p] done before next overwrite
        p ^= 1;
    }

    // epilogue: bias add + store (C layout: row = r + 8*(lane>=16), col = lane&15)
#pragma unroll
    for (int half = 0; half < 2; ++half) {
#pragma unroll
        for (int t = 0; t < 4; ++t) {
#pragma unroll
            for (int r = 0; r < 8; ++r) {
                int row = rowbase + half * 16 + r + ((lane >> 4) << 3);
                int col = colbase + t * 16 + (lane & 15);
                float v = acc[half][t][r] + bias[col];
                if (OUT_BF16) ((bf16*)Cout)[(size_t)row * Nout + col] = (bf16)v;
                else          ((float*)Cout)[(size_t)row * Nout + col] = v;
            }
        }
    }
}

// ---------------------------------------------------------------------------
// Flash-style attention. qkv: bf16 [B*N][3072], col = three*1024 + h*64 + d.
// Block = 8 waves; each wave owns 16 query rows; block covers 128 rows of one
// (b,h). KV chunks of 32: K async-DMA'd double-buffered, V transposed manually.
// ---------------------------------------------------------------------------
__global__ void __launch_bounds__(256)
attention_kernel(const bf16* __restrict__ qkv, bf16* __restrict__ ctx) {
    __shared__ __align__(16) bf16 Ks[2][32][64]; // [buf][kv][d]
    __shared__ __align__(16) bf16 Vt[64][32];    // [d][kv]
    __shared__ __align__(16) bf16 Ps[8][16][32]; // per-wave P tile [qrow][kv]

    const int tid  = threadIdx.x;
    const int lane = tid & 31;
    const int wave = tid >> 5;
    const int b = blockIdx.y >> 4;
    const int h = blockIdx.y & 15;
    const int qrow = blockIdx.x * 128 + wave * 16;

    const int ld = 3 * EMB;              // 3072
    const bf16* Qb = qkv + (size_t)b * SEQ * ld + h * HDIM;
    const bf16* Kb = Qb + EMB;
    const bf16* Vb = Qb + 2 * EMB;

    // Q fragments for this wave's 16 rows (resident across the whole KV loop)
    v16bf qa0 = load_frag_a(Qb, ld, qrow, 0,  lane);
    v16bf qa1 = load_frag_a(Qb, ld, qrow, 32, lane);

    // staging assignment: one b128 per thread per tile (32 kv x 64 d = 4KB)
    const int skr = tid >> 3;            // kv row 0..31
    const int sd  = (tid & 7) * 8;       // d 0..56
    const unsigned lds_elem = (unsigned)(skr * 128 + sd * 2);
    const unsigned lds_k0 = (unsigned)(size_t)(&Ks[0][0][0]) + lds_elem;
    const unsigned lds_k1 = (unsigned)(size_t)(&Ks[1][0][0]) + lds_elem;

    // prologue: DMA K tile 0 into buffer 0
    async_load_b128(lds_k0, (unsigned)((skr * ld + sd) * 2), Kb);

    v8f acc[4] = {};
    float mrow[8], lrow[8];
#pragma unroll
    for (int r = 0; r < 8; ++r) { mrow[r] = -1e30f; lrow[r] = 0.f; }

    int p = 0;
    for (int kv = 0; kv < SEQ; kv += 32) {
        // ---- stage V tile kv (VGPR transpose-scatter into Vt) ----
        {
            v8bf vv = *(const v8bf*)(Vb + (size_t)(kv + skr) * ld + sd);
#pragma unroll
            for (int i = 0; i < 8; ++i) Vt[sd + i][skr] = vv[i];
        }
        // ---- prefetch K tile kv+32 into alternate buffer, keep in flight ----
        if (kv + 32 < SEQ) {
            async_load_b128(p ? lds_k0 : lds_k1,
                            (unsigned)(((kv + 32 + skr) * ld + sd) * 2), Kb);
            wait_async1();          // wait only for tile kv (older, in-order)
        } else {
            wait_async0();
        }
        __syncthreads();            // K tile kv + V tile kv visible

        // ---- S = Q * K^T for 16x32 tile (two 16x16 WMMA tiles) ----
        v8f s[2];
#pragma unroll
        for (int c = 0; c < 2; ++c) {
            v8f z = {};
            v16bf bk0 = load_frag_b(&Ks[p][0][0], 64, c * 16, 0,  lane);
            v16bf bk1 = load_frag_b(&Ks[p][0][0], 64, c * 16, 32, lane);
            z = wmma_bf16(qa0, bk0, z);
            z = wmma_bf16(qa1, bk1, z);
            s[c] = z;
        }

        // ---- online softmax: per-row reductions across 16-lane row groups ----
#pragma unroll
        for (int r = 0; r < 8; ++r) {
            float s0 = s[0][r] * 0.125f;              // 1/sqrt(64)
            float s1 = s[1][r] * 0.125f;
            float mx = fmaxf(s0, s1);
#pragma unroll
            for (int off = 8; off >= 1; off >>= 1)
                mx = fmaxf(mx, __shfl_xor(mx, off, 32));
            float mn = fmaxf(mrow[r], mx);
            float sf = __expf(mrow[r] - mn);
            float p0 = __expf(s0 - mn);
            float p1 = __expf(s1 - mn);
            float ps = p0 + p1;
#pragma unroll
            for (int off = 8; off >= 1; off >>= 1)
                ps += __shfl_xor(ps, off, 32);
            lrow[r] = lrow[r] * sf + ps;
            mrow[r] = mn;
#pragma unroll
            for (int t = 0; t < 4; ++t) acc[t][r] *= sf;
            // stash P in LDS (C layout -> A layout crossing)
            int row = r + ((lane >> 4) << 3);
            Ps[wave][row][lane & 15]        = (bf16)p0;
            Ps[wave][row][16 + (lane & 15)] = (bf16)p1;
        }

        // ---- acc += P(16x32) * V(32x64) ---- (same-wave LDS ops are in order)
        v16bf pa = load_frag_a(&Ps[wave][0][0], 32, 0, 0, lane);
#pragma unroll
        for (int t = 0; t < 4; ++t) {
            v16bf bv = load_frag_b(&Vt[0][0], 32, t * 16, 0, lane);
            acc[t] = wmma_bf16(pa, bv, acc[t]);
        }
        __syncthreads();            // Ks[p]/Vt reads done before next overwrite
        p ^= 1;
    }

    // ---- normalize and write context: ctx[(b*N+n)][h*64 + d] (bf16) ----
#pragma unroll
    for (int t = 0; t < 4; ++t) {
#pragma unroll
        for (int r = 0; r < 8; ++r) {
            int n   = qrow + r + ((lane >> 4) << 3);
            int col = h * HDIM + t * 16 + (lane & 15);
            float v = acc[t][r] / lrow[r];
            ctx[((size_t)b * SEQ + n) * EMB + col] = (bf16)v;
        }
    }
}

// ---------------------------------------------------------------------------
// Launcher
// ---------------------------------------------------------------------------
extern "C" void kernel_launch(void* const* d_in, const int* in_sizes, int n_in,
                              void* d_out, int out_size, void* d_ws, size_t ws_size,
                              hipStream_t stream) {
    (void)in_sizes; (void)n_in; (void)out_size; (void)ws_size;
    const float* x      = (const float*)d_in[0];
    const float* w_qkv  = (const float*)d_in[1];
    const float* b_qkv  = (const float*)d_in[2];
    const float* w_proj = (const float*)d_in[3];
    const float* b_proj = (const float*)d_in[4];
    float* out = (float*)d_out;

    // workspace layout (bytes)
    char* ws = (char*)d_ws;
    bf16* xbf    = (bf16*)(ws + 0);                        // 8192*1024*2  = 16 MB
    bf16* wqkvT  = (bf16*)(ws + (size_t)16777216);         // 3072*1024*2  =  6 MB
    bf16* wprojT = (bf16*)(ws + (size_t)23068672);         // 1024*1024*2  =  2 MB
    bf16* qkvbf  = (bf16*)(ws + (size_t)25165824);         // 8192*3072*2  = 48 MB
    bf16* ctxbf  = (bf16*)(ws + (size_t)75497472);         // 8192*1024*2  = 16 MB

    const int nx4 = (MROWS * EMB) / 4;   // float4 elements
    cast_bf16_kernel<<<(nx4 + 255) / 256, 256, 0, stream>>>(x, xbf, nx4);
    transpose_cast_kernel<<<dim3(EMB / 256, 3 * EMB), 256, 0, stream>>>(w_qkv, wqkvT, EMB, 3 * EMB);
    transpose_cast_kernel<<<dim3(EMB / 256, EMB), 256, 0, stream>>>(w_proj, wprojT, EMB, EMB);

    // QKV projection: [8192 x 1024] * [1024 x 3072] -> bf16 qkv (+bias)
    gemm_bf16_kernel<true><<<dim3((3 * EMB) / 64, MROWS / 256), 256, 0, stream>>>(
        xbf, wqkvT, b_qkv, (void*)qkvbf, MROWS, 3 * EMB, EMB);

    // attention: grid (qblocks=16, B*H=64)
    attention_kernel<<<dim3(SEQ / 128, BATCH * HEADS), 256, 0, stream>>>(qkvbf, ctxbf);

    // output projection: [8192 x 1024] * [1024 x 1024] -> fp32 out (+bias)
    gemm_bf16_kernel<false><<<dim3(EMB / 64, MROWS / 256), 256, 0, stream>>>(
        ctxbf, wprojT, b_proj, (void*)out, MROWS, EMB, EMB);
}
